// SCARF_70317204570708
// MI455X (gfx1250) — matrix operand order repaired and checked by reading, not compiled, not run on hardware
//
#include <hip/hip_runtime.h>

typedef __attribute__((ext_vector_type(16))) _Float16 v16h;
typedef __attribute__((ext_vector_type(8)))  _Float16 v8h;
typedef __attribute__((ext_vector_type(8)))  float    v8f;
typedef __attribute__((ext_vector_type(4)))  unsigned int u32x4;
typedef __attribute__((ext_vector_type(8)))  int          i32x8;
typedef __attribute__((ext_vector_type(4)))  int          i32x4;

// async-to-LDS b128 builtin operand types (from hipcc diagnostic: param is
// "__attribute__((__vector_size__(4 * sizeof(int)))) int __device__ *", i.e.
// v4i in addrspace(1) / addrspace(3))
typedef int v4i_b128 __attribute__((vector_size(16)));
typedef __attribute__((address_space(1))) v4i_b128* async_gptr;
typedef __attribute__((address_space(3))) v4i_b128* async_lptr;

#define MDIM 512
#define HDIM 256
#define TILE_ROWS 64
#define LDSW_STRIDE 272                 // 256 K-halves + 16 pad -> 544B rows (32B aligned)
#define ACT_STRIDE  272
#define WCHUNK_HALVES (256 * LDSW_STRIDE)   // 69632 halves
#define ACT_HALVES    (64 * ACT_STRIDE)     // 17408 halves
#define LDS_BYTES ((WCHUNK_HALVES + 4 * ACT_HALVES) * 2)  // 278528 B <= 320KB WGP LDS

// --------- CDNA5 data-mover feature detection (compile-safe fallbacks) ------
#if defined(__gfx1250__) && __has_builtin(__builtin_amdgcn_tensor_load_to_lds) && (__clang_major__ < 23)
#define USE_TDM 1          // ROCm 7.2 clang-22: 5-arg tensor_load_to_lds
#endif
#if defined(__gfx1250__) && __has_builtin(__builtin_amdgcn_global_load_async_to_lds_b128)
#define USE_ASYNC_LDS 1    // per-lane async global->LDS (ASYNCcnt)
#endif

typedef __attribute__((address_space(3))) char lds_char;

__device__ __forceinline__ unsigned lds_addr_of(const void* p) {
  // generic -> LDS addrspace cast; integer value is the byte address in LDS
  return (unsigned)(unsigned long long)(lds_char*)p;
}

// ---------------------------------------------------------------------------
// Stage a [256][256]-half chunk of a transposed weight matrix (row-major
// Wt[N][Ksrc], columns c0..c0+255) into LDS with padded 272-half stride.
// ---------------------------------------------------------------------------
__device__ __forceinline__ void stage_w_fallback(_Float16* lds_w, const _Float16* __restrict__ wt,
                                                 int ksrc, int c0, int tid) {
  for (int idx = tid; idx < 256 * 32; idx += 256) {
    int n  = idx >> 5;
    int ko = (idx & 31) << 3;          // 8 halves = 16B per copy
    uint4 v = *(const uint4*)(wt + (size_t)n * ksrc + c0 + ko);
    *(uint4*)(lds_w + n * LDSW_STRIDE + ko) = v;
  }
}

#if defined(USE_TDM)
// One Tensor-Data-Mover descriptor moves the whole 256x256-half tile and
// inserts the 16-half row padding in hardware (D# pad fields).
__device__ __forceinline__ void stage_w_tdm(_Float16* lds_w, const _Float16* wt,
                                            int ksrc, int c0) {
  const unsigned long long ga = (unsigned long long)(const void*)(wt + (size_t)c0);
  u32x4 g0;
  g0[0] = 1u;                                     // count=1 (valid user D#)
  g0[1] = lds_addr_of(lds_w);                     // lds_addr (bytes)
  g0[2] = (unsigned)(ga & 0xffffffffu);           // global_addr[31:0]
  g0[3] = (unsigned)((ga >> 32) & 0x1ffffffu)     // global_addr[56:32]
        | (2u << 30);                             // type = 2 ("image")
  i32x8 g1;
  g1[0] = (1 << 16)                               // data_size = 2 bytes
        | (1 << 20)                               // pad_enable
        | (6 << 22)                               // pad_interval: 128 DW = 256 halves
        | (7 << 25);                              // pad_amount:   8 DW  = 16 halves
  g1[1] = (ksrc & 0xffff) << 16;                  // tensor_dim0[15:0] (K length)
  g1[2] = (256 << 16);                            // tensor_dim0 hi=0 | tensor_dim1 lo16 = 256 rows
  g1[3] = (256 << 16);                            // tensor_dim1 hi=0 | tile_dim0 = 256
  g1[4] = 256;                                    // tile_dim1 = 256, tile_dim2 = 0
  g1[5] = ksrc;                                   // tensor_dim0_stride lo32
  g1[6] = 0;
  g1[7] = 0;
  i32x4 g2 = {0, 0, 0, 0};                        // 2D tensor: groups 2/3 zero
  i32x4 g3 = {0, 0, 0, 0};
  __builtin_amdgcn_tensor_load_to_lds(g0, g1, g2, g3, 0);
}
#endif

__device__ __forceinline__ void stage_weights(_Float16* lds_w, const _Float16* __restrict__ wt,
                                              int ksrc, int c0, int tid) {
#if defined(USE_TDM)
  if (tid < 32) {                                  // one DMA per workgroup (wave 0)
    stage_w_tdm(lds_w, wt, ksrc, c0);
    __builtin_amdgcn_s_wait_tensorcnt(0);          // TENSORcnt == 0 -> tile in LDS
  }
#elif defined(USE_ASYNC_LDS)
  for (int idx = tid; idx < 256 * 32; idx += 256) {
    int n  = idx >> 5;
    int ko = (idx & 31) << 3;                      // 8 halves = 16B per lane transfer
    __builtin_amdgcn_global_load_async_to_lds_b128(
        (async_gptr)(wt + (size_t)n * ksrc + c0 + ko),
        (async_lptr)(lds_w + n * LDSW_STRIDE + ko), 0, 0);
  }
#if __has_builtin(__builtin_amdgcn_s_wait_asynccnt)
  __builtin_amdgcn_s_wait_asynccnt(0);
#else
  asm volatile("s_wait_asynccnt 0" ::: "memory");
#endif
#else
  stage_w_fallback(lds_w, wt, ksrc, c0, tid);
#endif
}

// A operand (16x32 f16): lane<16 holds row=lane, K 0..15;
// lane>=16 holds row=lane-16, K 16..31.
__device__ __forceinline__ v16h load_a(const _Float16* p /* row base + kk*32 */, int lane) {
  const int co = (lane >> 4) << 4;     // 0 or 16 halves
  v8h lo = *(const v8h*)(p + co);
  v8h hi = *(const v8h*)(p + 8 + co);
  return __builtin_shufflevector(lo, hi, 0, 1, 2, 3, 4, 5, 6, 7,
                                 8, 9, 10, 11, 12, 13, 14, 15);
}

__device__ __forceinline__ void zero_acc(v8f (&acc)[4][4]) {
  const v8f z = {0.f, 0.f, 0.f, 0.f, 0.f, 0.f, 0.f, 0.f};
#pragma unroll
  for (int mt = 0; mt < 4; ++mt)
#pragma unroll
    for (int nt = 0; nt < 4; ++nt) acc[mt][nt] = z;
}

// One K=256 GEMM slab: 4 M-tiles x 4 N-tiles per wave, A from asrc (global or
// LDS, stride in halves), B from staged LDS weights.
__device__ __forceinline__ void mm_k256(v8f (&acc)[4][4], const _Float16* asrc,
                                        int astride, const _Float16* lds_w,
                                        int n0, int lane) {
  const int ln    = lane & 15;
  const int khalf = (lane >> 4) << 4;  // B operand K offset (0 or 16 halves)
#pragma unroll
  for (int kk = 0; kk < 8; ++kk) {
    v16h a[4];
#pragma unroll
    for (int mt = 0; mt < 4; ++mt)
      a[mt] = load_a(asrc + (size_t)(mt * 16 + ln) * astride + kk * 32, lane);
#pragma unroll
    for (int nt = 0; nt < 4; ++nt) {
      v16h b = *(const v16h*)(lds_w + ((n0 + nt) * 16 + ln) * LDSW_STRIDE + kk * 32 + khalf);
#pragma unroll
      for (int mt = 0; mt < 4; ++mt)
        acc[mt][nt] = __builtin_amdgcn_wmma_f32_16x16x32_f16(
            false, a[mt], false, b, (short)0, acc[mt][nt], false, false);
    }
  }
}

// bias + optional relu, f32 acc -> f16 activation tile in LDS.
// C/D layout: vgpr i holds (M = i + 8*(lane>>4), N = lane&15).
__device__ __forceinline__ void store_act(v8f (&acc)[4][4], const float* __restrict__ bias,
                                          _Float16* dst, int n0, int lane, bool relu) {
  const int ln = lane & 15;
  const int mo = (lane >> 4) << 3;
#pragma unroll
  for (int mt = 0; mt < 4; ++mt)
#pragma unroll
    for (int nt = 0; nt < 4; ++nt) {
      const float bsc = bias[(n0 + nt) * 16 + ln];
#pragma unroll
      for (int i = 0; i < 8; ++i) {
        float v = acc[mt][nt][i] + bsc;
        if (relu) v = fmaxf(v, 0.0f);
        dst[(mt * 16 + mo + i) * ACT_STRIDE + (n0 + nt) * 16 + ln] = (_Float16)v;
      }
    }
}

__device__ __forceinline__ void hidden_layer(v8f (&acc)[4][4], _Float16* lds_w,
                                             const _Float16* src, _Float16* dst,
                                             const _Float16* __restrict__ wt,
                                             const float* __restrict__ bias, bool relu,
                                             int n0, int lane, int tid) {
  __syncthreads();                         // prior reads of lds_w / writes of dst done
  stage_weights(lds_w, wt, HDIM, 0, tid);
  __syncthreads();
  zero_acc(acc);
  mm_k256(acc, src, ACT_STRIDE, lds_w, n0, lane);
  store_act(acc, bias, dst, n0, lane, relu);
}

// ---------------------------------------------------------------------------
// Fused 6-layer MLP: block = 64 rows x both branches. Waves 0-3 clean branch,
// waves 4-7 corrupted branch; each wave owns 64 rows x 64 cols (4x4 tiles).
// ---------------------------------------------------------------------------
__global__ __launch_bounds__(256) void mlp_fused(
    const _Float16* __restrict__ xh, const _Float16* __restrict__ xch,
    const _Float16* __restrict__ wt_all,
    const float* __restrict__ b0, const float* __restrict__ b1,
    const float* __restrict__ b2, const float* __restrict__ b3,
    const float* __restrict__ b4, const float* __restrict__ b5,
    float* __restrict__ out, int Btot) {
  extern __shared__ _Float16 lds[];
  _Float16* lds_w = lds;

  const int tid    = threadIdx.x;
  const int wave   = tid >> 5;
  const int lane   = tid & 31;
  const int branch = wave >> 2;
  const int n0     = (wave & 3) << 2;       // first of this wave's 4 N-tiles
  const int ln     = lane & 15;
  const int row0   = blockIdx.x * TILE_ROWS;

  _Float16* act0 = lds + WCHUNK_HALVES + branch * 2 * ACT_HALVES;
  _Float16* act1 = act0 + ACT_HALVES;

  const _Float16* Xg = branch ? xch : xh;
  const _Float16* wt0 = wt_all;             // [256][512]
  const _Float16* wt1 = wt_all + 131072;    // [256][256] each below
  const _Float16* wt2 = wt_all + 196608;
  const _Float16* wt3 = wt_all + 262144;
  const _Float16* wt4 = wt_all + 327680;
  const _Float16* wt5 = wt_all + 393216;

  // warm caches for this block's layer-0 activation rows (global_prefetch_b8)
  {
    const char* xp = (const char*)(Xg + (size_t)row0 * MDIM);
    const int lid = tid & 127;
#pragma unroll
    for (int i = 0; i < 4; ++i)
      __builtin_prefetch(xp + (size_t)(lid * 4 + i) * 128, 0, 3);
  }

  v8f acc[4][4];
  zero_acc(acc);

  // ----- layer 0: K=512, A streamed from global f16, W staged in two halves
  __syncthreads();
  stage_weights(lds_w, wt0, MDIM, 0, tid);
  __syncthreads();
  mm_k256(acc, Xg + (size_t)row0 * MDIM, MDIM, lds_w, n0, lane);
  __syncthreads();
  stage_weights(lds_w, wt0, MDIM, 256, tid);
  __syncthreads();
  mm_k256(acc, Xg + (size_t)row0 * MDIM + 256, MDIM, lds_w, n0, lane);
  store_act(acc, b0, act0, n0, lane, true);

  // ----- layers 1..4 ping-pong through LDS
  hidden_layer(acc, lds_w, act0, act1, wt1, b1, true,  n0, lane, tid); // relu
  hidden_layer(acc, lds_w, act1, act0, wt2, b2, true,  n0, lane, tid); // relu
  hidden_layer(acc, lds_w, act0, act1, wt3, b3, false, n0, lane, tid); // encoder out
  hidden_layer(acc, lds_w, act1, act0, wt4, b4, true,  n0, lane, tid); // head relu

  // ----- layer 5: final, f32 output
  __syncthreads();
  stage_weights(lds_w, wt5, HDIM, 0, tid);
  __syncthreads();
  zero_acc(acc);
  mm_k256(acc, act0, ACT_STRIDE, lds_w, n0, lane);
  {
    const int mo = (lane >> 4) << 3;
    const size_t obase = (size_t)branch * Btot * HDIM;
#pragma unroll
    for (int mt = 0; mt < 4; ++mt)
#pragma unroll
      for (int nt = 0; nt < 4; ++nt) {
        const float bsc = b5[(n0 + nt) * 16 + ln];
#pragma unroll
        for (int i = 0; i < 8; ++i)
          out[obase + (size_t)(row0 + mt * 16 + mo + i) * HDIM + (n0 + nt) * 16 + ln] =
              acc[mt][nt][i] + bsc;
      }
  }
}

// ---------------------------------------------------------------------------
// Per-row corruption: rank(u) < q selects the q smallest u per row.
// One 512-thread block per row; emits x and x_corrupted as f16.
// ---------------------------------------------------------------------------
__global__ __launch_bounds__(512) void corrupt_rows(
    const float* __restrict__ x, const float* __restrict__ u, const float* __restrict__ r,
    const float* __restrict__ low, const float* __restrict__ high,
    const int* __restrict__ qp, _Float16* __restrict__ xh, _Float16* __restrict__ xch) {
  __shared__ float su[MDIM];
  const int b = blockIdx.x;
  const int m = threadIdx.x;
  const size_t off = (size_t)b * MDIM + m;
  const float um = u[off];
  su[m] = um;
  __syncthreads();
  int cnt = 0;
#pragma unroll 8
  for (int j = 0; j < MDIM; ++j) {
    const float uj = su[j];                // broadcast read
    cnt += (uj < um) || (uj == um && j < m);  // stable-sort tie break
  }
  const int q = qp[0];
  const float xv = x[off];
  const float xc = (cnt < q) ? (low[m] + (high[m] - low[m]) * r[off]) : xv;
  xh[off]  = (_Float16)xv;
  xch[off] = (_Float16)xc;
}

// W[K][N] f32 -> Wt[N][K] f16 (transposed so WMMA B loads are contiguous).
__global__ void wconv(const float* __restrict__ W, _Float16* __restrict__ Wt, int K, int N) {
  int idx = blockIdx.x * blockDim.x + threadIdx.x;
  if (idx >= K * N) return;
  int k = idx / N, n = idx - k * N;
  Wt[(size_t)n * K + k] = (_Float16)W[(size_t)k * N + n];
}

// ---------------------------------------------------------------------------
extern "C" void kernel_launch(void* const* d_in, const int* in_sizes, int n_in,
                              void* d_out, int out_size, void* d_ws, size_t ws_size,
                              hipStream_t stream) {
  const float* x    = (const float*)d_in[0];
  const float* u    = (const float*)d_in[1];
  const float* r    = (const float*)d_in[2];
  const float* low  = (const float*)d_in[3];
  const float* high = (const float*)d_in[4];
  const int*   q    = (const int*)d_in[5];
  const float* We0 = (const float*)d_in[6];  const float* be0 = (const float*)d_in[7];
  const float* We1 = (const float*)d_in[8];  const float* be1 = (const float*)d_in[9];
  const float* We2 = (const float*)d_in[10]; const float* be2 = (const float*)d_in[11];
  const float* We3 = (const float*)d_in[12]; const float* be3 = (const float*)d_in[13];
  const float* Wh0 = (const float*)d_in[14]; const float* bh0 = (const float*)d_in[15];
  const float* Wh1 = (const float*)d_in[16]; const float* bh1 = (const float*)d_in[17];
  float* out = (float*)d_out;

  const int B = in_sizes[0] / MDIM;          // 32768

  // workspace layout (halves): 6 transposed f16 weights, then xh, xch
  _Float16* ws  = (_Float16*)d_ws;
  _Float16* wt_all = ws;
  _Float16* wtp[6] = {ws, ws + 131072, ws + 196608, ws + 262144, ws + 327680, ws + 393216};
  _Float16* xh  = ws + 458752;
  _Float16* xch = xh + (size_t)B * MDIM;

  // 1) weight convert+transpose (tiny)
  wconv<<<(MDIM * HDIM + 255) / 256, 256, 0, stream>>>(We0, wtp[0], MDIM, HDIM);
  wconv<<<(HDIM * HDIM + 255) / 256, 256, 0, stream>>>(We1, wtp[1], HDIM, HDIM);
  wconv<<<(HDIM * HDIM + 255) / 256, 256, 0, stream>>>(We2, wtp[2], HDIM, HDIM);
  wconv<<<(HDIM * HDIM + 255) / 256, 256, 0, stream>>>(We3, wtp[3], HDIM, HDIM);
  wconv<<<(HDIM * HDIM + 255) / 256, 256, 0, stream>>>(Wh0, wtp[4], HDIM, HDIM);
  wconv<<<(HDIM * HDIM + 255) / 256, 256, 0, stream>>>(Wh1, wtp[5], HDIM, HDIM);

  // 2) rank mask + corruption + f16 conversion
  corrupt_rows<<<B, MDIM, 0, stream>>>(x, u, r, low, high, q, xh, xch);

  // 3) fused 6-layer MLP, both branches per block
  mlp_fused<<<B / TILE_ROWS, 256, LDS_BYTES, stream>>>(
      xh, xch, wt_all, be0, be1, be2, be3, bh0, bh1, out, B);
}
